// StdLeakLIF_68006512165368
// MI455X (gfx1250) — matrix operand order repaired and checked by reading, not compiled, not run on hardware
//
#include <hip/hip_runtime.h>

// ---- CDNA5 / gfx1250 fused LIF step: [x|z] @ [w_in;w_rec] via bf16 WMMA + fused epilogue
// Memory-bound (~448MB @ 23.3TB/s ~= 19us floor). bf16 WMMA w/ f32 accumulate keeps
// the matrix pipe far above the ~450 TF/s needed, so HBM stays the limiter.
// B-matrix weights are staged in LDS pre-shuffled into per-lane WMMA fragment order:
// all 128 ds_load_b128 in the hot loop share ONE address register + immediate offsets,
// and dual accumulators remove the WMMA->WMMA accumulator-hazard nops.

typedef __attribute__((ext_vector_type(16))) __bf16 v16bf;
typedef __attribute__((ext_vector_type(8)))  float  v8f;

#define NREC   128
#define KTOT   256
#define THRV   0.4f
#define LOG2_DECAY (-0.07400058144377693f)   // log2(1 - 1/20)

__device__ __forceinline__ v8f lif_wmma(v16bf a, v16bf b, v8f c) {
    return __builtin_amdgcn_wmma_f32_16x16x32_bf16(
        false, a, false, b, (short)0, c, false, false);
}

__global__ __launch_bounds__(256)
void lif_fused_wmma_kernel(const float* __restrict__ x,
                           const float* __restrict__ z,
                           const float* __restrict__ v,
                           const float* __restrict__ t,
                           const float* __restrict__ w_in,
                           const float* __restrict__ w_rec,
                           float* __restrict__ out_z,
                           float* __restrict__ out_v,
                           float* __restrict__ out_t) {
    // 64 B-fragments (8 N-tiles x 8 K-chunks), 512 bf16 (1KB) each = 64KB.
    // Fragment f layout: bytes [0,512): lane L elts 0..7 at L*16
    //                    bytes [512,1024): lane L elts 8..15 at 512 + L*16
    __shared__ __bf16 Wt[64 * 512];

    const int tid = threadIdx.x;

    // ---- stage weights into LDS in WMMA B-fragment order -------------------
    // consecutive tids read consecutive n (coalesced 512B rows of w), scatter to LDS.
    // B 32x16 bf16 fragment: lane L holds col n = nt*16 + L%16,
//   K = c*32 + (L/16)*8 + e for e<8 (low uint4), c*32 + 16 + (L/16)*8 + (e-8) for e>=8 (high uint4).
    for (int i = tid; i < 128 * KTOT; i += 256) {
        int k = i >> 7;                    // 0..255
        int n = i & 127;                   // 0..127
        float w = (k < 128) ? w_in[k * NREC + n] : w_rec[(k - 128) * NREC + n];
        int c   = k >> 5;                  // K chunk 0..7
        int k32 = k & 31;
        int L   = ((k32 >> 3) & 1) * 16 + (n & 15);
        int e   = (k32 & 7) | ((k32 >> 4) << 3);
        int f   = (n >> 4) * 8 + c;        // fragment id: n-tile major
        int dst = f * 512 + (e >> 3) * 256 + L * 8 + (e & 7);
        Wt[dst] = (__bf16)w;
    }
    __syncthreads();

    const int wid  = tid >> 5;             // wave id 0..7
    const int lane = tid & 31;
    const int l16  = lane & 15;
    const int hi   = lane >> 4;            // lane half

    const int rowBase = blockIdx.x * 128 + wid * 16;
    const int arow    = rowBase + l16;     // this lane's A-matrix row
    const float* xrow = x + (size_t)arow * NREC;
    const float* zrow = z + (size_t)arow * NREC;

    // ---- load A fragments: K=256 in 8 chunks of 32, f32 -> bf16 ------------
    // ISA 16-bit A 16x32 layout: elts 0..7  = K(kc + hi*8 + 0..7),
    //                            elts 8..15 = K(kc + 16 + hi*8 + 0..7)
    v16bf afrag[8];
#pragma unroll
    for (int c = 0; c < 8; ++c) {
        const int kc = c * 32;
        const float* src = (kc < 128) ? (xrow + kc) : (zrow + (kc - 128));
        const float* p = src + hi * 8;
        float4 f0 = *(const float4*)(p);
        float4 f1 = *(const float4*)(p + 4);
        float4 f2 = *(const float4*)(p + 16);
        float4 f3 = *(const float4*)(p + 20);
        v16bf a;
        a[0]  = (__bf16)f0.x; a[1]  = (__bf16)f0.y; a[2]  = (__bf16)f0.z; a[3]  = (__bf16)f0.w;
        a[4]  = (__bf16)f1.x; a[5]  = (__bf16)f1.y; a[6]  = (__bf16)f1.z; a[7]  = (__bf16)f1.w;
        a[8]  = (__bf16)f2.x; a[9]  = (__bf16)f2.y; a[10] = (__bf16)f2.z; a[11] = (__bf16)f2.w;
        a[12] = (__bf16)f3.x; a[13] = (__bf16)f3.y; a[14] = (__bf16)f3.z; a[15] = (__bf16)f3.w;
        afrag[c] = a;
    }

    // single shared LDS address register; every fragment is an immediate offset
    const __bf16* wbase = &Wt[lane * 8];   // lane*16 bytes

    // ---- 4 pairs of N-tiles, dual accumulators to break WMMA D->C hazards --
#pragma unroll
    for (int p = 0; p < 4; ++p) {
        v8f acc0 = {0.f, 0.f, 0.f, 0.f, 0.f, 0.f, 0.f, 0.f};
        v8f acc1 = {0.f, 0.f, 0.f, 0.f, 0.f, 0.f, 0.f, 0.f};

#pragma unroll
        for (int c = 0; c < 8; ++c) {
            const int f0 = (2 * p) * 8 + c;        // fragment ids (compile-time)
            const int f1 = (2 * p + 1) * 8 + c;
            union { v16bf bv; uint4 u[2]; } b0, b1;
            b0.u[0] = *(const uint4*)(wbase + f0 * 512);
            b0.u[1] = *(const uint4*)(wbase + f0 * 512 + 256);
            b1.u[0] = *(const uint4*)(wbase + f1 * 512);
            b1.u[1] = *(const uint4*)(wbase + f1 * 512 + 256);
            acc0 = lif_wmma(afrag[c], b0.bv, acc0);
            acc1 = lif_wmma(afrag[c], b1.bv, acc1);
        }

        // ---- fused LIF epilogue on both 16x16 i_in tiles --------------------
        // C/D layout: lane, VGPR r -> row = rowBase + hi*8 + r, col = n0 + l16
#pragma unroll
        for (int side = 0; side < 2; ++side) {
            const v8f acc = side ? acc1 : acc0;
            const int n0  = p * 32 + side * 16;
#pragma unroll
            for (int r = 0; r < 8; ++r) {
                const size_t idx = (size_t)(rowBase + hi * 8 + r) * NREC + (n0 + l16);
                float ii = acc[r];
                float zz = z[idx];
                float vv = __builtin_nontemporal_load(v + idx);
                float tt = __builtin_nontemporal_load(t + idx);

                float h   = (ii != 0.0f) ? 1.0f : 0.0f;
                float ntv = tt + 1.0f - h;                     // t + (h != 1)
                float nv  = vv * (1.0f - zz);                  // reset on spike
                nv = (vv > -1.0f) ? nv : (nv - (vv + 1.0f));   // clamp below -1
                nv = nv * __builtin_amdgcn_exp2f(h * (ntv + 1.0f) * LOG2_DECAY);
                ntv = ntv * (1.0f - h);                        // reset counter if active
                nv  = nv + ii;
                float nz = (nv > THRV) ? 1.0f : 0.0f;          // spike((nv-THR)/THR)

                __builtin_nontemporal_store(nz,  out_z + idx);
                __builtin_nontemporal_store(nv,  out_v + idx);
                __builtin_nontemporal_store(ntv, out_t + idx);
            }
        }
    }
}

extern "C" void kernel_launch(void* const* d_in, const int* in_sizes, int n_in,
                              void* d_out, int out_size, void* d_ws, size_t ws_size,
                              hipStream_t stream) {
    const float* x     = (const float*)d_in[0];
    const float* z     = (const float*)d_in[1];
    const float* v     = (const float*)d_in[2];
    const float* t     = (const float*)d_in[3];
    const float* w_in  = (const float*)d_in[4];
    const float* w_rec = (const float*)d_in[5];

    const int Brows = in_sizes[0] / NREC;          // 131072
    float* out   = (float*)d_out;                  // [new_z | new_v | new_t]
    float* out_z = out;
    float* out_v = out + (size_t)Brows * NREC;
    float* out_t = out + 2 * (size_t)Brows * NREC;

    const int blocks = Brows / 128;                // 128 rows per block (8 waves x 16)
    lif_fused_wmma_kernel<<<dim3(blocks), dim3(256), 0, stream>>>(
        x, z, v, t, w_in, w_rec, out_z, out_v, out_t);
}
